// NodeModel_32813550141461
// MI455X (gfx1250) — matrix operand order repaired and checked by reading, not compiled, not run on hardware
//
#include <hip/hip_runtime.h>
#include <hip/hip_bf16.h>

#define NN 50000
#define EE 800000
#define BN_EPS 1e-5f

typedef __attribute__((ext_vector_type(16))) __bf16 v16bf;
typedef __attribute__((ext_vector_type(8)))  float  v8f;

union Frag { v16bf v; uint4 q[2]; };

__device__ __forceinline__ v8f wmma_bf16(const Frag& a, const Frag& b, v8f c) {
  return __builtin_amdgcn_wmma_f32_16x16x32_bf16(false, a.v, false, b.v,
                                                 (short)0, c, false, false);
}
// A fragment: 16x32 bf16 tile, lane row = lane&15, K halves split by lane>>4
__device__ __forceinline__ void loadA(Frag& f, const __bf16* a_k0, int kbA) {
  f.q[0] = *(const uint4*)(a_k0 + 2 * kbA);
  f.q[1] = *(const uint4*)(a_k0 + 2 * kbA + 8);
}
// B fragment from col-major weights: lane col = lane&15, K half = (lane>>4)*16
__device__ __forceinline__ void loadB(Frag& f, const __bf16* b_k0, int kbB) {
  f.q[0] = *(const uint4*)(b_k0 + kbB);
  f.q[1] = *(const uint4*)(b_k0 + kbB + 8);
}

// ---------------------------------------------------------------- zero scratch
__global__ void zero_ws(float* __restrict__ p, long long n) {
  long long i = (long long)blockIdx.x * blockDim.x + threadIdx.x;
  if (i < n) p[i] = 0.f;
}

// ---------------------------------------------------------------- edge GEMM1
// A = [x[send] | edge_attr]  (E x 128) @ w1m (128 x 64) + b1m
// out: h1 (bf16, E x 64) + column sum/sumsq stats (f32)
__global__ __launch_bounds__(256) void edge_gemm1(
    const float* __restrict__ x, const int* __restrict__ ei,
    const float* __restrict__ ea, const float* __restrict__ w1,
    const float* __restrict__ b1, __bf16* __restrict__ h1,
    float* __restrict__ stats)
{
  __shared__ __bf16 Wt[64 * 128];     // col-major: Wt[n][k]
  __shared__ __bf16 As[128 * 128];
  __shared__ __bf16 Hs[128 * 64];
  __shared__ float s_sum[64], s_ssq[64];
  const int tid = threadIdx.x;
  const long long e0 = (long long)blockIdx.x * 128;

  for (int i = tid; i < 128 * 64; i += 256) {
    int k = i >> 6, n = i & 63;
    Wt[n * 128 + k] = (__bf16)w1[i];
  }
  if (tid < 64) { s_sum[tid] = 0.f; s_ssq[tid] = 0.f; }
#pragma unroll
  for (int rr = 0; rr < 4; ++rr) {
    int row = rr * 32 + (tid >> 3);
    int cb  = (tid & 7) * 16;
    long long e = e0 + row;
    const float* src = (cb < 64) ? (x + (long long)ei[e] * 64 + cb)
                                 : (ea + e * 64 + (cb - 64));
    const float4* s4 = (const float4*)src;
    __bf16* dst = As + row * 128 + cb;
#pragma unroll
    for (int j = 0; j < 4; ++j) {
      float4 v = s4[j];
      dst[j*4+0] = (__bf16)v.x; dst[j*4+1] = (__bf16)v.y;
      dst[j*4+2] = (__bf16)v.z; dst[j*4+3] = (__bf16)v.w;
    }
  }
  __syncthreads();

  const int lane = tid & 31, wave = tid >> 5;
  const int colb = lane & 15, half = lane >> 4;
  const int kbA = half * 8, kbB = half * 16;
  const int row0 = wave * 16 + half * 8;
  const __bf16* arow = As + (wave * 16 + colb) * 128;

  v8f acc[4];
#pragma unroll
  for (int ct = 0; ct < 4; ++ct) {
    float b = b1[ct * 16 + colb];
#pragma unroll
    for (int i = 0; i < 8; ++i) acc[ct][i] = b;
  }
#pragma unroll
  for (int k0 = 0; k0 < 128; k0 += 32) {
    Frag a; loadA(a, arow + k0, kbA);
#pragma unroll
    for (int ct = 0; ct < 4; ++ct) {
      Frag b; loadB(b, Wt + (ct * 16 + colb) * 128 + k0, kbB);
      acc[ct] = wmma_bf16(a, b, acc[ct]);
    }
  }
#pragma unroll
  for (int ct = 0; ct < 4; ++ct) {
    int col = ct * 16 + colb;
    float s = 0.f, ss = 0.f;
#pragma unroll
    for (int i = 0; i < 8; ++i) {
      float v = acc[ct][i];
      s += v; ss += v * v;
      Hs[(row0 + i) * 64 + col] = (__bf16)v;
    }
    atomicAdd(&s_sum[col], s);
    atomicAdd(&s_ssq[col], ss);
  }
  __syncthreads();
  {
    uint4* dst = (uint4*)(h1 + e0 * 64);
    const uint4* src = (const uint4*)Hs;
    dst[tid * 2]     = src[tid * 2];
    dst[tid * 2 + 1] = src[tid * 2 + 1];
  }
  if (tid < 64) {
    atomicAdd(&stats[tid], s_sum[tid]);
    atomicAdd(&stats[64 + tid], s_ssq[tid]);
  }
}

// ---------------------------------------------------------------- BN finalize
__global__ void bn_finalize(const float* __restrict__ stats,
                            const float* __restrict__ gamma,
                            const float* __restrict__ beta,
                            float* __restrict__ out, float inv_count)
{
  int i = threadIdx.x;            // 64 threads
  float mean = stats[i] * inv_count;
  float var  = stats[64 + i] * inv_count - mean * mean;
  float sc = gamma[i] * rsqrtf(var + BN_EPS);
  out[i]      = sc;
  out[64 + i] = beta[i] - mean * sc;
}

// ---------------------------------------------------------------- edge GEMM2 + scatter
__global__ __launch_bounds__(256) void edge_gemm2_scatter(
    const __bf16* __restrict__ h1, const int* __restrict__ ei,
    const float* __restrict__ w2, const float* __restrict__ b2,
    const float* __restrict__ ss, float* __restrict__ nsum,
    float* __restrict__ ncnt)
{
  __shared__ __bf16 Wt[64 * 64];
  __shared__ __bf16 As[128 * 64];
  __shared__ float s_sc[64], s_sh[64];
  __shared__ int recs[128];
  const int tid = threadIdx.x;
  const long long e0 = (long long)blockIdx.x * 128;

  for (int i = tid; i < 64 * 64; i += 256) {
    int k = i >> 6, n = i & 63;
    Wt[n * 64 + k] = (__bf16)w2[i];
  }
  if (tid < 64) { s_sc[tid] = ss[tid]; s_sh[tid] = ss[64 + tid]; }
  if (tid < 128) recs[tid] = ei[EE + e0 + tid];
  union { uint4 q; __bf16 h[8]; } u0, u1;
  u0.q = ((const uint4*)(h1 + e0 * 64))[tid * 2];
  u1.q = ((const uint4*)(h1 + e0 * 64))[tid * 2 + 1];
  __syncthreads();
  {
    int row = tid >> 2, cb = (tid & 3) * 16;
#pragma unroll
    for (int j = 0; j < 8; ++j) {
      float f = (float)u0.h[j] * s_sc[cb + j] + s_sh[cb + j];
      As[row * 64 + cb + j] = (__bf16)fmaxf(f, 0.f);
      float g = (float)u1.h[j] * s_sc[cb + 8 + j] + s_sh[cb + 8 + j];
      As[row * 64 + cb + 8 + j] = (__bf16)fmaxf(g, 0.f);
    }
  }
  __syncthreads();

  const int lane = tid & 31, wave = tid >> 5;
  const int colb = lane & 15, half = lane >> 4;
  const int kbA = half * 8, kbB = half * 16;
  const int row0 = wave * 16 + half * 8;
  const __bf16* arow = As + (wave * 16 + colb) * 64;

  v8f acc[4];
#pragma unroll
  for (int ct = 0; ct < 4; ++ct) {
    float b = b2[ct * 16 + colb];
#pragma unroll
    for (int i = 0; i < 8; ++i) acc[ct][i] = b;
  }
#pragma unroll
  for (int k0 = 0; k0 < 64; k0 += 32) {
    Frag a; loadA(a, arow + k0, kbA);
#pragma unroll
    for (int ct = 0; ct < 4; ++ct) {
      Frag b; loadB(b, Wt + (ct * 16 + colb) * 64 + k0, kbB);
      acc[ct] = wmma_bf16(a, b, acc[ct]);
    }
  }
#pragma unroll
  for (int ct = 0; ct < 4; ++ct) {
    int col = ct * 16 + colb;
#pragma unroll
    for (int i = 0; i < 8; ++i) {
      int rdst = recs[row0 + i];
      atomicAdd(&nsum[(long long)rdst * 64 + col], acc[ct][i]);
    }
  }
  if (tid < 128) atomicAdd(&ncnt[recs[tid]], 1.0f);
}

// ---------------------------------------------------------------- node GEMM1
__global__ __launch_bounds__(256) void node_gemm1(
    const float* __restrict__ x, const float* __restrict__ nsum,
    const float* __restrict__ ncnt, const float* __restrict__ w1,
    const float* __restrict__ b1, __bf16* __restrict__ h2,
    float* __restrict__ stats)
{
  __shared__ __bf16 Wt[64 * 128];
  __shared__ __bf16 As[128 * 128];
  __shared__ __bf16 Hs[128 * 64];
  __shared__ float s_sum[64], s_ssq[64];
  const int tid = threadIdx.x;
  const long long n0 = (long long)blockIdx.x * 128;

  for (int i = tid; i < 128 * 64; i += 256) {
    int k = i >> 6, n = i & 63;
    Wt[n * 128 + k] = (__bf16)w1[i];
  }
  if (tid < 64) { s_sum[tid] = 0.f; s_ssq[tid] = 0.f; }
#pragma unroll
  for (int rr = 0; rr < 4; ++rr) {
    int row = rr * 32 + (tid >> 3);
    int cb  = (tid & 7) * 16;
    long long n = n0 + row;
    __bf16* dst = As + row * 128 + cb;
    if (n < NN) {
      if (cb < 64) {
        const float4* s4 = (const float4*)(x + n * 64 + cb);
#pragma unroll
        for (int j = 0; j < 4; ++j) {
          float4 v = s4[j];
          dst[j*4+0] = (__bf16)v.x; dst[j*4+1] = (__bf16)v.y;
          dst[j*4+2] = (__bf16)v.z; dst[j*4+3] = (__bf16)v.w;
        }
      } else {
        float inv = 1.0f / fmaxf(ncnt[n], 1.0f);
        const float4* s4 = (const float4*)(nsum + n * 64 + (cb - 64));
#pragma unroll
        for (int j = 0; j < 4; ++j) {
          float4 v = s4[j];
          dst[j*4+0] = (__bf16)(v.x * inv); dst[j*4+1] = (__bf16)(v.y * inv);
          dst[j*4+2] = (__bf16)(v.z * inv); dst[j*4+3] = (__bf16)(v.w * inv);
        }
      }
    } else {
#pragma unroll
      for (int j = 0; j < 16; ++j) dst[j] = (__bf16)0.f;
    }
  }
  __syncthreads();

  const int lane = tid & 31, wave = tid >> 5;
  const int colb = lane & 15, half = lane >> 4;
  const int kbA = half * 8, kbB = half * 16;
  const int row0 = wave * 16 + half * 8;
  const __bf16* arow = As + (wave * 16 + colb) * 128;

  v8f acc[4];
#pragma unroll
  for (int ct = 0; ct < 4; ++ct) {
    float b = b1[ct * 16 + colb];
#pragma unroll
    for (int i = 0; i < 8; ++i) acc[ct][i] = b;
  }
#pragma unroll
  for (int k0 = 0; k0 < 128; k0 += 32) {
    Frag a; loadA(a, arow + k0, kbA);
#pragma unroll
    for (int ct = 0; ct < 4; ++ct) {
      Frag b; loadB(b, Wt + (ct * 16 + colb) * 128 + k0, kbB);
      acc[ct] = wmma_bf16(a, b, acc[ct]);
    }
  }
#pragma unroll
  for (int ct = 0; ct < 4; ++ct) {
    int col = ct * 16 + colb;
    float s = 0.f, ss = 0.f;
#pragma unroll
    for (int i = 0; i < 8; ++i) {
      bool valid = (n0 + row0 + i) < NN;
      float v = valid ? acc[ct][i] : 0.f;
      s += v; ss += v * v;
      Hs[(row0 + i) * 64 + col] = (__bf16)acc[ct][i];
    }
    atomicAdd(&s_sum[col], s);
    atomicAdd(&s_ssq[col], ss);
  }
  __syncthreads();
  {
    int orow = tid >> 2;          // two uint4 per thread, same h2 row
    if (n0 + orow < NN) {
      uint4* dst = (uint4*)(h2 + n0 * 64);
      const uint4* src = (const uint4*)Hs;
      dst[tid * 2]     = src[tid * 2];
      dst[tid * 2 + 1] = src[tid * 2 + 1];
    }
  }
  if (tid < 64) {
    atomicAdd(&stats[tid], s_sum[tid]);
    atomicAdd(&stats[64 + tid], s_ssq[tid]);
  }
}

// ---------------------------------------------------------------- node GEMM2 -> out
__global__ __launch_bounds__(256) void node_gemm2_out(
    const __bf16* __restrict__ h2, const float* __restrict__ w2,
    const float* __restrict__ b2, const float* __restrict__ ss,
    float* __restrict__ out)
{
  __shared__ __bf16 Wt[64 * 64];
  __shared__ __bf16 As[128 * 64];
  __shared__ float Outs[128 * 64];
  __shared__ float s_sc[64], s_sh[64];
  const int tid = threadIdx.x;
  const long long n0 = (long long)blockIdx.x * 128;

  for (int i = tid; i < 64 * 64; i += 256) {
    int k = i >> 6, n = i & 63;
    Wt[n * 64 + k] = (__bf16)w2[i];
  }
  if (tid < 64) { s_sc[tid] = ss[tid]; s_sh[tid] = ss[64 + tid]; }
  union { uint4 q; __bf16 h[8]; } u0, u1;
  {
    int row = tid >> 2;
    if (n0 + row < NN) {
      u0.q = ((const uint4*)(h2 + n0 * 64))[tid * 2];
      u1.q = ((const uint4*)(h2 + n0 * 64))[tid * 2 + 1];
    } else {
      u0.q = make_uint4(0u, 0u, 0u, 0u);
      u1.q = make_uint4(0u, 0u, 0u, 0u);
    }
  }
  __syncthreads();
  {
    int row = tid >> 2, cb = (tid & 3) * 16;
#pragma unroll
    for (int j = 0; j < 8; ++j) {
      float f = (float)u0.h[j] * s_sc[cb + j] + s_sh[cb + j];
      As[row * 64 + cb + j] = (__bf16)fmaxf(f, 0.f);
      float g = (float)u1.h[j] * s_sc[cb + 8 + j] + s_sh[cb + 8 + j];
      As[row * 64 + cb + 8 + j] = (__bf16)fmaxf(g, 0.f);
    }
  }
  __syncthreads();

  const int lane = tid & 31, wave = tid >> 5;
  const int colb = lane & 15, half = lane >> 4;
  const int kbA = half * 8, kbB = half * 16;
  const int row0 = wave * 16 + half * 8;
  const __bf16* arow = As + (wave * 16 + colb) * 64;

  v8f acc[4];
#pragma unroll
  for (int ct = 0; ct < 4; ++ct) {
    float b = b2[ct * 16 + colb];
#pragma unroll
    for (int i = 0; i < 8; ++i) acc[ct][i] = b;
  }
#pragma unroll
  for (int k0 = 0; k0 < 64; k0 += 32) {
    Frag a; loadA(a, arow + k0, kbA);
#pragma unroll
    for (int ct = 0; ct < 4; ++ct) {
      Frag b; loadB(b, Wt + (ct * 16 + colb) * 64 + k0, kbB);
      acc[ct] = wmma_bf16(a, b, acc[ct]);
    }
  }
#pragma unroll
  for (int ct = 0; ct < 4; ++ct) {
    int col = ct * 16 + colb;
#pragma unroll
    for (int i = 0; i < 8; ++i) Outs[(row0 + i) * 64 + col] = acc[ct][i];
  }
  __syncthreads();
  {
    int row = tid >> 1;           // 8 float4 per thread, one h-row each
    if (n0 + row < NN) {
      float4* dst = (float4*)(out + n0 * 64);
      const float4* src = (const float4*)Outs;
#pragma unroll
      for (int j = 0; j < 8; ++j) dst[tid * 8 + j] = src[tid * 8 + j];
    }
  }
}

// ---------------------------------------------------------------- launch
extern "C" void kernel_launch(void* const* d_in, const int* in_sizes, int n_in,
                              void* d_out, int out_size, void* d_ws, size_t ws_size,
                              hipStream_t stream)
{
  const float* x   = (const float*)d_in[0];
  const int*   ei  = (const int*)d_in[1];
  const float* ea  = (const float*)d_in[2];
  const float* w1m = (const float*)d_in[5];
  const float* b1m = (const float*)d_in[6];
  const float* gm  = (const float*)d_in[7];
  const float* btm = (const float*)d_in[8];
  const float* w2m = (const float*)d_in[9];
  const float* b2m = (const float*)d_in[10];
  const float* w1n = (const float*)d_in[11];
  const float* b1n = (const float*)d_in[12];
  const float* gn  = (const float*)d_in[13];
  const float* btn = (const float*)d_in[14];
  const float* w2n = (const float*)d_in[15];
  const float* b2n = (const float*)d_in[16];
  float* out = (float*)d_out;

  float* ws      = (float*)d_ws;
  float* stats_m = ws;                 // 128 f32
  float* ssm     = ws + 128;           // 128 f32 (scale|shift)
  float* stats_n = ws + 256;           // 128 f32
  float* ssn     = ws + 384;           // 128 f32
  float* ncnt    = ws + 512;           // N f32
  float* nsum    = ws + 512 + NN;      // N*64 f32
  __bf16* h1 = (__bf16*)(ws + 512 + NN + (size_t)NN * 64);  // E*64 bf16
  __bf16* h2 = h1 + (size_t)EE * 64;                        // N*64 bf16

  long long nz = 512 + NN + (long long)NN * 64;
  zero_ws<<<(unsigned)((nz + 255) / 256), 256, 0, stream>>>(ws, nz);

  edge_gemm1<<<EE / 128, 256, 0, stream>>>(x, ei, ea, w1m, b1m, h1, stats_m);
  bn_finalize<<<1, 64, 0, stream>>>(stats_m, gm, btm, ssm, 1.0f / (float)EE);
  edge_gemm2_scatter<<<EE / 128, 256, 0, stream>>>(h1, ei, w2m, b2m, ssm, nsum, ncnt);
  node_gemm1<<<(NN + 127) / 128, 256, 0, stream>>>(x, nsum, ncnt, w1n, b1n, h2, stats_n);
  bn_finalize<<<1, 64, 0, stream>>>(stats_n, gn, btn, ssn, 1.0f / (float)NN);
  node_gemm2_out<<<(NN + 127) / 128, 256, 0, stream>>>(h2, w2n, b2n, ssn, out);
}
